// CoAtten2_68289980006551
// MI455X (gfx1250) — matrix operand order repaired and checked by reading, not compiled, not run on hardware
//
#include <hip/hip_runtime.h>

// ---------------------------------------------------------------------------
// CoAtten2 for MI455X (gfx1250): bf16 WMMA + TDM pipeline.
// C=1024, H=W=64, HW=4096.
//   Qc = Wq@Xm+bq ; Kf = Wk1@Xf+bk1 ; Kl = Wk2@Xl+bk2   -> each [512,4096]
//   (the torch .view(1024,2048) of a contiguous [512,4096] buffer is simply
//    the same buffer reinterpreted row-major [1024,2048])
//   V = Wv@Xm+bv [1024,4096]
//   attF = softmax(KFv @ Qv^T), attL = softmax(KLv @ Qv^T)   (NT, [1024,1024])
//   out  = gamma*((attF+attL)@V) + 0.5*(x_f+x_l)
// ---------------------------------------------------------------------------

typedef __attribute__((ext_vector_type(16))) __bf16 v16bf;
typedef __attribute__((ext_vector_type(8)))  __bf16 v8bf;
typedef __attribute__((ext_vector_type(8)))  float  v8f;
typedef __attribute__((ext_vector_type(4)))  unsigned int v4u;
typedef __attribute__((ext_vector_type(8)))  int v8i;
typedef __attribute__((ext_vector_type(4)))  int v4i;

__device__ __forceinline__ __bf16 f2bf(float f) {
    unsigned u = __builtin_bit_cast(unsigned, f);
    u += 0x7fffu + ((u >> 16) & 1u);                 // round-to-nearest-even
    unsigned short h = (unsigned short)(u >> 16);
    return __builtin_bit_cast(__bf16, h);
}

// ---------------------------------------------------------------------------
// Tensor Data Mover: DMA a [rows x 32] bf16 tile (row stride = ld elements)
// from global memory into LDS at byte offset lds_off.  D# built per the
// CDNA5 ISA group0/group1 layouts (data_size=2B, 2D tile, no pad/iterate).
// This toolchain exposes the 6-arg builtin (g0, g1, g2, g3, g_extra, cpol).
// ---------------------------------------------------------------------------
__device__ __forceinline__ void tdm_load_tile_2d(unsigned lds_off,
                                                 const __bf16* gptr,
                                                 unsigned tile_k, unsigned rows,
                                                 unsigned ld) {
    unsigned long long ga = (unsigned long long)(const void*)gptr;
    v4u g0;
    g0.x = 1u;                                            // count=1, user mode
    g0.y = lds_off;                                       // lds_addr (bytes)
    g0.z = (unsigned)ga;                                  // global_addr[31:0]
    g0.w = (unsigned)((ga >> 32) & 0x01ffffffu) | (2u << 30); // addr[56:32], type=2
    unsigned td0 = ld;            // tensor_dim0 (elements) -> tile always in range
    unsigned td1 = 1u << 20;      // tensor_dim1: large, tile always in range
    v8i g1 = {};
    g1[0] = (int)(1u << 16);                              // data_size=1 -> 2 bytes
    g1[1] = (int)((td0 & 0xffffu) << 16);                 // tensor_dim0[15:0]
    g1[2] = (int)(((td0 >> 16) & 0xffffu) | ((td1 & 0xffffu) << 16));
    g1[3] = (int)(((td1 >> 16) & 0xffffu) | (tile_k << 16)); // tile_dim0
    g1[4] = (int)rows;                                    // tile_dim1
    g1[5] = (int)ld;                                      // tensor_dim0_stride[31:0]
    g1[6] = 0;                                            // stride[47:32], dim1_stride lo
    g1[7] = 0;
    v4i z4 = {0, 0, 0, 0};
    v8i z8 = {};
    __builtin_amdgcn_tensor_load_to_lds(g0, g1, z4, z4, z8, 0);
}

// ---------------------------------------------------------------------------
// f32 -> bf16 convert (n multiple of 4)
// ---------------------------------------------------------------------------
__global__ void cvt_f32_bf16(const float* __restrict__ in, __bf16* __restrict__ out, int n) {
    int i = (blockIdx.x * blockDim.x + threadIdx.x) * 4;
    if (i + 3 < n) {
        float4 f = *reinterpret_cast<const float4*>(in + i);
        out[i + 0] = f2bf(f.x);
        out[i + 1] = f2bf(f.y);
        out[i + 2] = f2bf(f.z);
        out[i + 3] = f2bf(f.w);
    }
}

// ---------------------------------------------------------------------------
// Tiled bf16 WMMA GEMM.  Block tile BM=128, BN=128, BK=32. 256 thr = 8 waves
// (4 in M x 2 in N), wave tile 32x64 = 2x4 fragments -> 8 WMMAs / K-tile.
// A tile (and B tile when NT) staged by the Tensor Data Mover; NN B tiles are
// loaded coalesced and transposed into LDS by the whole block.
// MODE 0 (CONV):  C_bf16 = A[M,K]@B[K,N] + bias[M]            (NN, ldb=N)
// MODE 1 (LOGIT): C_f32  = A[M,K]@B[N,K]^T                    (NT, ldb=K)
// MODE 2 (FINAL): C_f32  = gamma*(A@B) + 0.5*(xf+xl)          (NN, ldb=N)
// ---------------------------------------------------------------------------
enum { MODE_CONV = 0, MODE_LOGIT = 1, MODE_FINAL = 2 };

template <int MODE, bool NT>
__global__ void __launch_bounds__(256)
gemm_wmma(const __bf16* __restrict__ A, const __bf16* __restrict__ B,
          int M, int N, int K, int ldb,
          const float* __restrict__ bias,
          __bf16* __restrict__ Obf,
          float* __restrict__ Of,
          const float* __restrict__ xf, const float* __restrict__ xl,
          const float* __restrict__ gamma) {
    __shared__ __bf16 As[128 * 32];   // [m][k]
    __shared__ __bf16 Bs[128 * 32];   // [n][k] (K-contiguous for B fragments)

    const int tid   = threadIdx.x;
    const int tileM = blockIdx.y * 128;
    const int tileN = blockIdx.x * 128;
    const int w     = tid >> 5;
    const int lane  = tid & 31;
    const int waveM = w >> 1;         // 0..3
    const int waveN = w & 1;          // 0..1
    const int lr    = lane & 15;
    const int hi    = lane >> 4;

    const unsigned asOff = (unsigned)(size_t)(void*)As;
    const unsigned bsOff = (unsigned)(size_t)(void*)Bs;

    v8f acc[2][4] = {};

    for (int k0 = 0; k0 < K; k0 += 32) {
        // ---- stage A (always) and B (NT) tiles via the Tensor Data Mover ----
        if (tid < 32) {
            tdm_load_tile_2d(asOff, A + (size_t)tileM * K + k0, 32, 128, (unsigned)K);
            if (NT)
                tdm_load_tile_2d(bsOff, B + (size_t)tileN * ldb + k0, 32, 128, (unsigned)ldb);
            __builtin_amdgcn_s_wait_tensorcnt(0);
        }
        // ---- NN: B is [K,N] row-major; coalesced load + transpose into LDS ----
        if (!NT) {
#pragma unroll
            for (int s = 0; s < 2; ++s) {
                int slot = tid + s * 256;       // 32 k-rows x 16 n-groups
                int krow = slot >> 4;
                int n8   = (slot & 15) * 8;
                v8bf v = *reinterpret_cast<const v8bf*>(
                    B + (size_t)(k0 + krow) * ldb + tileN + n8);
#pragma unroll
                for (int j = 0; j < 8; ++j) Bs[(n8 + j) * 32 + krow] = v[j];
            }
            if (k0 + 32 < K)
                __builtin_prefetch(B + (size_t)(k0 + 32 + (tid >> 4)) * ldb +
                                       tileN + (tid & 15) * 8, 0, 1);
        }
        __syncthreads();

        // ---- fragments per the CDNA5 16-bit WMMA VGPR layouts ----
        v16bf afrag[2], bfrag[4];
#pragma unroll
        for (int mf = 0; mf < 2; ++mf) {
            int row = waveM * 32 + mf * 16 + lr;
            v8bf lo  = *reinterpret_cast<const v8bf*>(&As[row * 32 + hi * 8]);
            v8bf hi8 = *reinterpret_cast<const v8bf*>(&As[row * 32 + hi * 8 + 16]);
            afrag[mf] = __builtin_shufflevector(lo, hi8, 0, 1, 2, 3, 4, 5, 6, 7,
                                                8, 9, 10, 11, 12, 13, 14, 15);
        }
#pragma unroll
        for (int nf = 0; nf < 4; ++nf) {
            int col = waveN * 64 + nf * 16 + lr;
            v8bf lo  = *reinterpret_cast<const v8bf*>(&Bs[col * 32 + hi * 8]);
            v8bf hi8 = *reinterpret_cast<const v8bf*>(&Bs[col * 32 + hi * 8 + 16]);
            bfrag[nf] = __builtin_shufflevector(lo, hi8, 0, 1, 2, 3, 4, 5, 6, 7,
                                                8, 9, 10, 11, 12, 13, 14, 15);
        }
#pragma unroll
        for (int mf = 0; mf < 2; ++mf)
#pragma unroll
            for (int nf = 0; nf < 4; ++nf)
                acc[mf][nf] = __builtin_amdgcn_wmma_f32_16x16x32_bf16(
                    false, afrag[mf], false, bfrag[nf], (short)0, acc[mf][nf],
                    false, false);
        __syncthreads();
    }

    // ---- epilogue (C/D layout: lane<16 -> M=r, lane>=16 -> M=8+r; N=lane&15) ----
    float g = 0.0f;
    if (MODE == MODE_FINAL) g = gamma[0];
#pragma unroll
    for (int mf = 0; mf < 2; ++mf) {
#pragma unroll
        for (int nf = 0; nf < 4; ++nf) {
            int gcol = tileN + waveN * 64 + nf * 16 + lr;
#pragma unroll
            for (int r = 0; r < 8; ++r) {
                int grow = tileM + waveM * 32 + mf * 16 + hi * 8 + r;
                size_t idx = (size_t)grow * N + gcol;
                float v = acc[mf][nf][r];
                if (MODE == MODE_CONV) {
                    Obf[idx] = f2bf(v + bias[grow]);
                } else if (MODE == MODE_LOGIT) {
                    Of[idx] = v;
                } else {
                    Of[idx] = g * v + 0.5f * (xf[idx] + xl[idx]);
                }
            }
        }
    }
}

// ---------------------------------------------------------------------------
// In-place row softmax: 1024 cols per row, 256 threads (8 waves) per row.
// ---------------------------------------------------------------------------
__global__ void softmax_rows(float* __restrict__ d, int ncols) {
    const int row = blockIdx.x, tid = threadIdx.x;
    float* p = d + (size_t)row * ncols;
    __shared__ float red[8];

    float v[4];
    float m = -3.4e38f;
#pragma unroll
    for (int i = 0; i < 4; ++i) { v[i] = p[tid + i * 256]; m = fmaxf(m, v[i]); }
#pragma unroll
    for (int o = 16; o > 0; o >>= 1) m = fmaxf(m, __shfl_xor(m, o, 32));
    if ((tid & 31) == 0) red[tid >> 5] = m;
    __syncthreads();
    float bm = red[0];
#pragma unroll
    for (int i = 1; i < 8; ++i) bm = fmaxf(bm, red[i]);
    __syncthreads();

    float s = 0.0f;
#pragma unroll
    for (int i = 0; i < 4; ++i) { v[i] = __expf(v[i] - bm); s += v[i]; }
#pragma unroll
    for (int o = 16; o > 0; o >>= 1) s += __shfl_xor(s, o, 32);
    if ((tid & 31) == 0) red[tid >> 5] = s;
    __syncthreads();
    float bs = 0.0f;
#pragma unroll
    for (int i = 0; i < 8; ++i) bs += red[i];
    float inv = 1.0f / bs;
#pragma unroll
    for (int i = 0; i < 4; ++i) p[tid + i * 256] = v[i] * inv;
}

__global__ void add_att_bf16(const float* __restrict__ a, const float* __restrict__ b,
                             __bf16* __restrict__ o, int n) {
    int i = blockIdx.x * blockDim.x + threadIdx.x;
    if (i < n) o[i] = f2bf(a[i] + b[i]);
}

// ---------------------------------------------------------------------------
extern "C" void kernel_launch(void* const* d_in, const int* in_sizes, int n_in,
                              void* d_out, int out_size, void* d_ws, size_t ws_size,
                              hipStream_t stream) {
    (void)in_sizes; (void)n_in; (void)out_size; (void)ws_size;
    const float* x_f  = (const float*)d_in[0];
    const float* x_m  = (const float*)d_in[1];
    const float* x_l  = (const float*)d_in[2];
    const float* Wq   = (const float*)d_in[3];
    const float* bq   = (const float*)d_in[4];
    const float* Wk1  = (const float*)d_in[5];
    const float* bk1  = (const float*)d_in[6];
    const float* Wk2  = (const float*)d_in[7];
    const float* bk2  = (const float*)d_in[8];
    const float* Wv   = (const float*)d_in[9];
    const float* bv   = (const float*)d_in[10];
    const float* gam  = (const float*)d_in[11];
    float* out = (float*)d_out;

    const int CHW   = 1024 * 4096;
    const int WHALF = 512 * 1024;
    const int WFULL = 1024 * 1024;

    char* ws = (char*)d_ws;
    size_t off = 0;
    auto alloc = [&](size_t bytes) {
        void* p = ws + off;
        off += (bytes + 255) & ~(size_t)255;
        return p;
    };
    __bf16* XmB  = (__bf16*)alloc((size_t)CHW * 2);
    __bf16* XfB  = (__bf16*)alloc((size_t)CHW * 2);
    __bf16* XlB  = (__bf16*)alloc((size_t)CHW * 2);
    __bf16* WqB  = (__bf16*)alloc((size_t)WHALF * 2);
    __bf16* Wk1B = (__bf16*)alloc((size_t)WHALF * 2);
    __bf16* Wk2B = (__bf16*)alloc((size_t)WHALF * 2);
    __bf16* WvB  = (__bf16*)alloc((size_t)WFULL * 2);
    __bf16* Qc   = (__bf16*)alloc((size_t)512 * 4096 * 2);   // == [1024,2048] view
    __bf16* Kf   = (__bf16*)alloc((size_t)512 * 4096 * 2);
    __bf16* Kl   = (__bf16*)alloc((size_t)512 * 4096 * 2);
    __bf16* Vb   = (__bf16*)alloc((size_t)1024 * 4096 * 2);
    float*  attF = (float*)alloc((size_t)1024 * 1024 * 4);
    float*  attL = (float*)alloc((size_t)1024 * 1024 * 4);
    __bf16* attS = (__bf16*)alloc((size_t)1024 * 1024 * 2);

    // 1) convert inputs + weights to bf16
    cvt_f32_bf16<<<CHW / 4 / 256, 256, 0, stream>>>(x_m, XmB, CHW);
    cvt_f32_bf16<<<CHW / 4 / 256, 256, 0, stream>>>(x_f, XfB, CHW);
    cvt_f32_bf16<<<CHW / 4 / 256, 256, 0, stream>>>(x_l, XlB, CHW);
    cvt_f32_bf16<<<WHALF / 4 / 256, 256, 0, stream>>>(Wq, WqB, WHALF);
    cvt_f32_bf16<<<WHALF / 4 / 256, 256, 0, stream>>>(Wk1, Wk1B, WHALF);
    cvt_f32_bf16<<<WHALF / 4 / 256, 256, 0, stream>>>(Wk2, Wk2B, WHALF);
    cvt_f32_bf16<<<WFULL / 4 / 256, 256, 0, stream>>>(Wv, WvB, WFULL);

    // 2) 1x1 conv GEMMs (NN + bias, bf16 out)
    gemm_wmma<MODE_CONV, false><<<dim3(32, 4), 256, 0, stream>>>(
        WqB, XmB, 512, 4096, 1024, 4096, bq, Qc, nullptr, nullptr, nullptr, nullptr);
    gemm_wmma<MODE_CONV, false><<<dim3(32, 4), 256, 0, stream>>>(
        Wk1B, XfB, 512, 4096, 1024, 4096, bk1, Kf, nullptr, nullptr, nullptr, nullptr);
    gemm_wmma<MODE_CONV, false><<<dim3(32, 4), 256, 0, stream>>>(
        Wk2B, XlB, 512, 4096, 1024, 4096, bk2, Kl, nullptr, nullptr, nullptr, nullptr);
    gemm_wmma<MODE_CONV, false><<<dim3(32, 8), 256, 0, stream>>>(
        WvB, XmB, 1024, 4096, 1024, 4096, bv, Vb, nullptr, nullptr, nullptr, nullptr);

    // 3) attention logits: NT GEMMs on the contiguous [1024,2048] views, f32 out
    gemm_wmma<MODE_LOGIT, true><<<dim3(8, 8), 256, 0, stream>>>(
        Kf, Qc, 1024, 1024, 2048, 2048, nullptr, nullptr, attF, nullptr, nullptr, nullptr);
    gemm_wmma<MODE_LOGIT, true><<<dim3(8, 8), 256, 0, stream>>>(
        Kl, Qc, 1024, 1024, 2048, 2048, nullptr, nullptr, attL, nullptr, nullptr, nullptr);

    // 4) row softmax (in place), then sum -> bf16
    softmax_rows<<<1024, 256, 0, stream>>>(attF, 1024);
    softmax_rows<<<1024, 256, 0, stream>>>(attL, 1024);
    add_att_bf16<<<(1024 * 1024) / 256, 256, 0, stream>>>(attF, attL, attS, 1024 * 1024);

    // 5) final GEMM + fused epilogue: gamma*(attS@V) + 0.5*(x_f+x_l)
    gemm_wmma<MODE_FINAL, false><<<dim3(32, 8), 256, 0, stream>>>(
        attS, Vb, 1024, 4096, 1024, 4096, nullptr, nullptr, out, x_f, x_l, gam);
}
